// LemLocal_24927990186015
// MI455X (gfx1250) — matrix operand order repaired and checked by reading, not compiled, not run on hardware
//
#include <hip/hip_runtime.h>
#include <hip/hip_bf16.h>

typedef __attribute__((ext_vector_type(16))) _Float16 v16h;
typedef __attribute__((ext_vector_type(8)))  _Float16 v8h;
typedef __attribute__((ext_vector_type(4)))  _Float16 v4h;
typedef __attribute__((ext_vector_type(8)))  float    v8f;

#define HDIM 128
#define EADIM 32
#define CCH 4
#define KPAD 320    // 292 padded to 10 K-blocks of 32 (weight layout)
#define KPADL 328   // LDS row stride for 320-col input (bank-conflict padding)
#define HSTR 136    // LDS row stride for 128-col buffers (bank-conflict padding)
#define TT 2        // edge M-tiles per wave (B-fragment reuse factor)

__device__ __forceinline__ float fast_rcp(float x) {
    return __builtin_amdgcn_rcpf(x);
}
__device__ __forceinline__ float silu_f(float x) {
    return x * fast_rcp(1.0f + __expf(-x));
}

// ---------------------------------------------------------------------------
// CDNA5 async global->LDS copy (16B per lane), ASYNCcnt-tracked.
// Raw ISA form (CDNA5 ISA): per-lane LDS address in the
// vdst slot, 64-bit global address, immediate offset applied to both sides.
// ---------------------------------------------------------------------------
__device__ __forceinline__ void async_ld_b128(const _Float16* g, _Float16* l, int off)
{
    unsigned laddr = (unsigned)(size_t)l;   // low 32 bits of flat-shared = LDS offset
    asm volatile("global_load_async_to_lds_b128 %0, %1, off offset:%2"
                 :: "v"(laddr), "v"(g), "i"(off) : "memory");
}
__device__ __forceinline__ void wait_async0()
{
    asm volatile("s_wait_asynccnt 0" ::: "memory");
}

// ---------------------------------------------------------------------------
// T-tile MLP partial: acc[t][16x16] = bias + LDS_in[t][16,K] @ Wpacked[K,16nb]
// A fragment (16-bit A 16x32 layout): lane hi=L>>4, row m=L&15:
//   halves 0..7  = K = kb*32 + hi*8 .. +7
//   halves 8..15 = K = kb*32 + 16 + hi*8 .. +7
// B fragment pre-packed: 32 lanes x 16 contiguous halves per (kb,nb) block,
// shared across all T tiles.
// C/D: element (M = j + 8*hi, N = nb*16 + (L&15)).
// ---------------------------------------------------------------------------
template <int KB, int NB, int T>
__device__ __forceinline__ void wmma_multi(const _Float16* lds, int tileStride,
                                           int stride,
                                           const _Float16* __restrict__ Wp,
                                           int nb, float bias, int lane, v8f* acc)
{
#pragma unroll
    for (int t = 0; t < T; ++t)
#pragma unroll
        for (int j = 0; j < 8; ++j) acc[t][j] = bias;
    const int m  = lane & 15;
    const int hi = lane >> 4;
#pragma unroll
    for (int kb = 0; kb < KB; ++kb) {
        v16h b = *(const v16h*)(Wp + (((size_t)kb * NB + nb) * 32 + lane) * 16);
#pragma unroll
        for (int t = 0; t < T; ++t) {
            const _Float16* base = lds + t * tileStride + m * stride + kb * 32;
            v8h a0 = *(const v8h*)(base + hi * 8);
            v8h a1 = *(const v8h*)(base + 16 + hi * 8);
            v16h a;
#pragma unroll
            for (int q = 0; q < 8; ++q) { a[q] = a0[q]; a[q + 8] = a1[q]; }
            acc[t] = __builtin_amdgcn_wmma_f32_16x16x32_f16(false, a, false, b,
                                                            (short)0, acc[t],
                                                            false, false);
        }
    }
}

// ---------------------------------------------------------------------------
// Pack fp32 weight W[Krows,Ncols] into B-fragment order; optionally place the
// bias vector at K row `biasRow` (bias rides free inside the K padding).
// ---------------------------------------------------------------------------
__global__ void pack_w_kernel(const float* __restrict__ W, const float* __restrict__ bias,
                              _Float16* __restrict__ out,
                              int Krows, int Ncols, int KB, int NB, int biasRow)
{
    int idx = blockIdx.x * blockDim.x + threadIdx.x;
    int total = KB * NB * 512;
    if (idx >= total) return;
    int h    = idx & 15;
    int lane = (idx >> 4) & 31;
    int blk  = idx >> 9;
    int nb = blk % NB, kb = blk / NB;
    int k = kb * 32 + (lane >> 4) * 8 + (h & 7) + (h >> 3) * 16;
    int n = nb * 16 + (lane & 15);
    float v = 0.0f;
    if (n < Ncols) {
        if (k < Krows)         v = W[(size_t)k * Ncols + n];
        else if (k == biasRow) v = bias[n];
    }
    out[idx] = (_Float16)v;
}

__global__ void zero_f32_kernel(float* __restrict__ p, size_t n)
{
    size_t i = (size_t)blockIdx.x * blockDim.x + threadIdx.x;
    size_t stride = (size_t)gridDim.x * blockDim.x;
    for (; i < n; i += stride) p[i] = 0.0f;
}

__global__ void cvt_feat_kernel(const float* __restrict__ in, _Float16* __restrict__ out,
                                size_t n4)
{
    size_t i = (size_t)blockIdx.x * blockDim.x + threadIdx.x;
    size_t stride = (size_t)gridDim.x * blockDim.x;
    for (; i < n4; i += stride) {
        float4 v = ((const float4*)in)[i];
        v4h h; h[0] = (_Float16)v.x; h[1] = (_Float16)v.y;
               h[2] = (_Float16)v.z; h[3] = (_Float16)v.w;
        ((v4h*)out)[i] = h;
    }
}

// ---------------------------------------------------------------------------
// Node init: vn_feat = silu(silu(x@W1+b1)@W2+b2)   (fi MLP, last_act=True)
// ---------------------------------------------------------------------------
__global__ __launch_bounds__(32)
void node_fi_kernel(const _Float16* __restrict__ feat16,
                    const _Float16* __restrict__ W1p, const float* __restrict__ b1,
                    const _Float16* __restrict__ W2p, const float* __restrict__ b2,
                    float* __restrict__ outFeat, int N)
{
    alignas(32) __shared__ _Float16 xin[16 * HSTR];
    alignas(32) __shared__ _Float16 hmid[16 * HSTR];
    const int L  = threadIdx.x;
    const int n0 = blockIdx.x * 16;

    {
        int r = L >> 1, half = L & 1;
        int n = n0 + r;
        bool ok = n < N;
        const v8h* src = (const v8h*)(feat16 + (size_t)(ok ? n : 0) * HDIM) + half * 8;
        _Float16* dst = xin + r * HSTR + half * 64;
#pragma unroll
        for (int i = 0; i < 8; ++i) {
            v8h v = src[i];
            if (!ok) {
#pragma unroll
                for (int q = 0; q < 8; ++q) v[q] = (_Float16)0.0f;
            }
            *(v8h*)(dst + i * 8) = v;
        }
    }
    __syncthreads();

    const int col = L & 15, hi = L >> 4;
    v8f acc[1];
    for (int nb = 0; nb < 8; ++nb) {
        wmma_multi<4, 8, 1>(xin, 0, HSTR, W1p, nb, b1[nb * 16 + col], L, acc);
#pragma unroll
        for (int j = 0; j < 8; ++j)
            hmid[(j + 8 * hi) * HSTR + nb * 16 + col] = (_Float16)silu_f(acc[0][j]);
    }
    __syncthreads();
    for (int nb = 0; nb < 8; ++nb) {
        wmma_multi<4, 8, 1>(hmid, 0, HSTR, W2p, nb, b2[nb * 16 + col], L, acc);
#pragma unroll
        for (int j = 0; j < 8; ++j) {
            int n = n0 + j + 8 * hi;
            if (n < N) outFeat[(size_t)n * HDIM + nb * 16 + col] = silu_f(acc[0][j]);
        }
    }
}

// ---------------------------------------------------------------------------
// Node finalize: vn_feat += vf(agg_feat / max(cnt,1))  (layer1 silu, layer2 none)
// ---------------------------------------------------------------------------
__global__ __launch_bounds__(32)
void node_vf_kernel(const float* __restrict__ aggF, const float* __restrict__ cnt,
                    const _Float16* __restrict__ W1p, const float* __restrict__ b1,
                    const _Float16* __restrict__ W2p, const float* __restrict__ b2,
                    float* __restrict__ outFeat, int N)
{
    alignas(32) __shared__ _Float16 xin[16 * HSTR];
    alignas(32) __shared__ _Float16 hmid[16 * HSTR];
    const int L  = threadIdx.x;
    const int n0 = blockIdx.x * 16;

    {
        int r = L >> 1, half = L & 1;
        int n = n0 + r;
        bool ok = n < N;
        float rs = ok ? fast_rcp(fmaxf(cnt[n], 1.0f)) : 0.0f;
        const float4* src = (const float4*)(aggF + (size_t)(ok ? n : 0) * HDIM) + half * 16;
        _Float16* dst = xin + r * HSTR + half * 64;
#pragma unroll
        for (int i = 0; i < 16; ++i) {
            float4 v = src[i];
            v4h h;
            h[0] = (_Float16)(v.x * rs);
            h[1] = (_Float16)(v.y * rs);
            h[2] = (_Float16)(v.z * rs);
            h[3] = (_Float16)(v.w * rs);
            *(v4h*)(dst + i * 4) = h;
        }
    }
    __syncthreads();

    const int col = L & 15, hi = L >> 4;
    v8f acc[1];
    for (int nb = 0; nb < 8; ++nb) {
        wmma_multi<4, 8, 1>(xin, 0, HSTR, W1p, nb, b1[nb * 16 + col], L, acc);
#pragma unroll
        for (int j = 0; j < 8; ++j)
            hmid[(j + 8 * hi) * HSTR + nb * 16 + col] = (_Float16)silu_f(acc[0][j]);
    }
    __syncthreads();
    for (int nb = 0; nb < 8; ++nb) {
        wmma_multi<4, 8, 1>(hmid, 0, HSTR, W2p, nb, b2[nb * 16 + col], L, acc);
#pragma unroll
        for (int j = 0; j < 8; ++j) {
            int n = n0 + j + 8 * hi;
            if (n < N) {
                size_t o = (size_t)n * HDIM + nb * 16 + col;
                outFeat[o] = outFeat[o] + acc[0][j];
            }
        }
    }
}

// ---------------------------------------------------------------------------
// Fused edge pipeline, TT 16-edge tiles per wave:
// async f16 gather -> concat(293 incl. bias-one, pad 320) -> msg MLP -> co MLP
// -> scatter-add (agg_feat, agg_vec, cnt).
// ---------------------------------------------------------------------------
__global__ __launch_bounds__(32)
void edge_kernel(const _Float16* __restrict__ feat16, const float* __restrict__ pos,
                 const int* __restrict__ eidx, const float* __restrict__ eattr,
                 const _Float16* __restrict__ mW1p,
                 const _Float16* __restrict__ mW2p, const float* __restrict__ mb2,
                 const _Float16* __restrict__ cW1p, const float* __restrict__ cb1,
                 const _Float16* __restrict__ cW2p, const float* __restrict__ cb2,
                 float* __restrict__ aggF, float* __restrict__ aggV,
                 float* __restrict__ cnt, int E)
{
    alignas(32) __shared__ _Float16 ein[TT * 16 * KPADL];
    alignas(32) __shared__ _Float16 hbuf[TT * 16 * HSTR];
    alignas(32) __shared__ _Float16 mbuf[TT * 16 * HSTR];
    __shared__ float sU[TT * 16][3];
    __shared__ int   sRow[TT * 16];
    __shared__ int   sCol[TT * 16];

    const int L = threadIdx.x;
    const long e0 = (long)blockIdx.x * (TT * 16);

    {
        long e = e0 + L;
        bool val = e < E;
        int r = val ? eidx[e]     : 0;
        int c = val ? eidx[E + e] : 0;
        sRow[L] = r; sCol[L] = c;

        _Float16* rowp = ein + (L >> 4) * (16 * KPADL) + (L & 15) * KPADL;

        // feat16[row] -> cols [0,128), feat16[col] -> cols [128,256):
        // DMA straight into LDS (ASYNCcnt), overlaps the work below.
        const _Float16* gr = feat16 + (size_t)r * HDIM;
        const _Float16* gc = feat16 + (size_t)c * HDIM;
#pragma unroll
        for (int i = 0; i < 16; ++i) async_ld_b128(gr, rowp, i * 16);
#pragma unroll
        for (int i = 0; i < 16; ++i) async_ld_b128(gc, rowp + HDIM, i * 16);

        float dx = pos[r * 3 + 0] - pos[c * 3 + 0];
        float dy = pos[r * 3 + 1] - pos[c * 3 + 1];
        float dz = pos[r * 3 + 2] - pos[c * 3 + 2];
        float dsq = dx * dx + dy * dy + dz * dz;
        float rn  = fast_rcp(fmaxf(sqrtf(dsq), 1e-8f));
        sU[L][0] = dx * rn; sU[L][1] = dy * rn; sU[L][2] = dz * rn;

        // edge_attr -> cols [256,288)
        const float4* fa = (const float4*)(eattr + (size_t)(val ? e : 0) * EADIM);
#pragma unroll
        for (int i = 0; i < 8; ++i) {
            float4 v = fa[i];
            v4h h; h[0] = (_Float16)v.x; h[1] = (_Float16)v.y;
                   h[2] = (_Float16)v.z; h[3] = (_Float16)v.w;
            *(v4h*)(rowp + 2 * HDIM + i * 4) = h;
        }
        // ip (4x dist_sq) -> [288,292); bias-one -> 292; zero pad -> (292,320)
        _Float16 ipv = (_Float16)dsq;
        v4h hip; hip[0] = ipv; hip[1] = ipv; hip[2] = ipv; hip[3] = ipv;
        *(v4h*)(rowp + 2 * HDIM + EADIM) = hip;
        v4h hb1; hb1[0] = (_Float16)1.0f; hb1[1] = hb1[2] = hb1[3] = (_Float16)0.0f;
        *(v4h*)(rowp + 2 * HDIM + EADIM + 4) = hb1;
        v4h hz; hz[0] = hz[1] = hz[2] = hz[3] = (_Float16)0.0f;
#pragma unroll
        for (int q = 2 * HDIM + EADIM + 8; q < KPAD; q += 4)
            *(v4h*)(rowp + q) = hz;
    }
    wait_async0();
    __syncthreads();

    const int col = L & 15, hi = L >> 4;
    v8f acc[TT];

    // --- msg layer 1: h = silu(in @ mW1'), bias folded into K row 292 ---
    for (int nb = 0; nb < 8; ++nb) {
        wmma_multi<10, 8, TT>(ein, 16 * KPADL, KPADL, mW1p, nb, 0.0f, L, acc);
#pragma unroll
        for (int t = 0; t < TT; ++t)
#pragma unroll
            for (int j = 0; j < 8; ++j)
                hbuf[t * 16 * HSTR + (j + 8 * hi) * HSTR + nb * 16 + col] =
                    (_Float16)silu_f(acc[t][j]);
    }
    __syncthreads();

    // --- msg layer 2: msg = h @ mW2 + mb2 ; scatter-add to agg_feat ---
    for (int nb = 0; nb < 8; ++nb) {
        wmma_multi<4, 8, TT>(hbuf, 16 * HSTR, HSTR, mW2p, nb,
                             mb2[nb * 16 + col], L, acc);
#pragma unroll
        for (int t = 0; t < TT; ++t)
#pragma unroll
            for (int j = 0; j < 8; ++j) {
                int M = t * 16 + j + 8 * hi;
                float x = acc[t][j];
                mbuf[t * 16 * HSTR + (j + 8 * hi) * HSTR + nb * 16 + col] = (_Float16)x;
                long e = e0 + M;
                if (e < E)
                    atomicAdd(&aggF[(size_t)sRow[M] * HDIM + nb * 16 + col], x);
            }
    }
    __syncthreads();

    // --- co layer 1: coh = silu(msg @ cW1 + cb1) (reuse hbuf) ---
    for (int nb = 0; nb < 8; ++nb) {
        wmma_multi<4, 8, TT>(mbuf, 16 * HSTR, HSTR, cW1p, nb,
                             cb1[nb * 16 + col], L, acc);
#pragma unroll
        for (int t = 0; t < TT; ++t)
#pragma unroll
            for (int j = 0; j < 8; ++j)
                hbuf[t * 16 * HSTR + (j + 8 * hi) * HSTR + nb * 16 + col] =
                    (_Float16)silu_f(acc[t][j]);
    }
    __syncthreads();

    // --- co layer 2: w_coff = coh @ cW2 + cb2 (N padded 4->16); scatter vec ---
    {
        float b = (col < CCH) ? cb2[col] : 0.0f;
        wmma_multi<4, 1, TT>(hbuf, 16 * HSTR, HSTR, cW2p, 0, b, L, acc);
        if (col < CCH) {
#pragma unroll
            for (int t = 0; t < TT; ++t)
#pragma unroll
                for (int j = 0; j < 8; ++j) {
                    int M = t * 16 + j + 8 * hi;
                    long e = e0 + M;
                    if (e < E) {
                        float w = acc[t][j];
                        size_t base = (size_t)sRow[M] * (3 * CCH) + col * 3;
                        atomicAdd(&aggV[base + 0], w * sU[M][0]);
                        atomicAdd(&aggV[base + 1], w * sU[M][1]);
                        atomicAdd(&aggV[base + 2], w * sU[M][2]);
                    }
                }
        }
    }

    // --- edge counts ---
    {
        long e = e0 + L;
        if (e < E) atomicAdd(&cnt[sRow[L]], 1.0f);
    }
}

// ---------------------------------------------------------------------------
// vn_pos = tile(pos, C) + agg_vec / max(cnt,1)
// ---------------------------------------------------------------------------
__global__ void pos_kernel(const float* __restrict__ pos, const float* __restrict__ aggV,
                           const float* __restrict__ cnt, float* __restrict__ outPos, int N)
{
    int idx = blockIdx.x * blockDim.x + threadIdx.x;
    if (idx >= N * 3 * CCH) return;
    int n = idx / (3 * CCH), q = idx % (3 * CCH), d = q % 3;
    float rs = fast_rcp(fmaxf(cnt[n], 1.0f));
    outPos[idx] = pos[(size_t)n * 3 + d] + aggV[idx] * rs;
}

// ---------------------------------------------------------------------------
extern "C" void kernel_launch(void* const* d_in, const int* in_sizes, int n_in,
                              void* d_out, int out_size, void* d_ws, size_t ws_size,
                              hipStream_t stream)
{
    (void)n_in; (void)out_size; (void)ws_size;
    const float* node_feat = (const float*)d_in[0];
    const float* node_pos  = (const float*)d_in[1];
    const int*   eidx      = (const int*)  d_in[2];
    const float* eattr     = (const float*)d_in[3];
    const float* fi_w1 = (const float*)d_in[4],  *fi_b1 = (const float*)d_in[5];
    const float* fi_w2 = (const float*)d_in[6],  *fi_b2 = (const float*)d_in[7];
    const float* mg_w1 = (const float*)d_in[8],  *mg_b1 = (const float*)d_in[9];
    const float* mg_w2 = (const float*)d_in[10], *mg_b2 = (const float*)d_in[11];
    const float* co_w1 = (const float*)d_in[12], *co_b1 = (const float*)d_in[13];
    const float* co_w2 = (const float*)d_in[14], *co_b2 = (const float*)d_in[15];
    const float* vf_w1 = (const float*)d_in[16], *vf_b1 = (const float*)d_in[17];
    const float* vf_w2 = (const float*)d_in[18], *vf_b2 = (const float*)d_in[19];

    const int N = in_sizes[0] / HDIM;
    const int E = in_sizes[2] / 2;

    char* ws = (char*)d_ws;
    size_t o = 0;
    auto carve = [&](size_t bytes) -> void* {
        void* p = ws + o;
        o += (bytes + 255) & ~(size_t)255;
        return p;
    };
    _Float16* fiW1p = (_Float16*)carve((size_t)4 * 8 * 512 * 2);
    _Float16* fiW2p = (_Float16*)carve((size_t)4 * 8 * 512 * 2);
    _Float16* mgW1p = (_Float16*)carve((size_t)10 * 8 * 512 * 2);
    _Float16* mgW2p = (_Float16*)carve((size_t)4 * 8 * 512 * 2);
    _Float16* coW1p = (_Float16*)carve((size_t)4 * 8 * 512 * 2);
    _Float16* coW2p = (_Float16*)carve((size_t)4 * 1 * 512 * 2);
    _Float16* vfW1p = (_Float16*)carve((size_t)4 * 8 * 512 * 2);
    _Float16* vfW2p = (_Float16*)carve((size_t)4 * 8 * 512 * 2);
    _Float16* feat16 = (_Float16*)carve((size_t)N * HDIM * 2);
    char* zeroBase = ws + o;
    float* aggF = (float*)carve((size_t)N * HDIM * 4);
    float* aggV = (float*)carve((size_t)N * 3 * CCH * 4);
    float* cnt  = (float*)carve((size_t)N * 4);
    size_t zeroFloats = ((ws + o) - zeroBase) / 4;

    auto pack = [&](const float* W, const float* bias, _Float16* out,
                    int K, int Nc, int KB, int NB, int biasRow) {
        int total = KB * NB * 512;
        pack_w_kernel<<<(total + 255) / 256, 256, 0, stream>>>(W, bias, out, K, Nc,
                                                               KB, NB, biasRow);
    };
    pack(fi_w1, fi_b1, fiW1p, HDIM, HDIM, 4, 8, -1);
    pack(fi_w2, fi_b2, fiW2p, HDIM, HDIM, 4, 8, -1);
    pack(mg_w1, mg_b1, mgW1p, 2 * HDIM + EADIM + 4, HDIM, 10, 8, 2 * HDIM + EADIM + 4);
    pack(mg_w2, mg_b2, mgW2p, HDIM, HDIM, 4, 8, -1);
    pack(co_w1, co_b1, coW1p, HDIM, HDIM, 4, 8, -1);
    pack(co_w2, co_b2, coW2p, HDIM, CCH, 4, 1, -1);
    pack(vf_w1, vf_b1, vfW1p, HDIM, HDIM, 4, 8, -1);
    pack(vf_w2, vf_b2, vfW2p, HDIM, HDIM, 4, 8, -1);

    cvt_feat_kernel<<<2048, 256, 0, stream>>>(node_feat, feat16,
                                              (size_t)N * HDIM / 4);
    zero_f32_kernel<<<2048, 256, 0, stream>>>((float*)zeroBase, zeroFloats);

    float* outFeat = (float*)d_out;
    float* outPos  = outFeat + (size_t)N * HDIM;

    node_fi_kernel<<<(N + 15) / 16, 32, 0, stream>>>(
        feat16, fiW1p, fi_b1, fiW2p, fi_b2, outFeat, N);

    edge_kernel<<<(E + TT * 16 - 1) / (TT * 16), 32, 0, stream>>>(
        feat16, node_pos, eidx, eattr,
        mgW1p, mgW2p, mg_b2,
        coW1p, co_b1, coW2p, co_b2,
        aggF, aggV, cnt, E);

    node_vf_kernel<<<(N + 15) / 16, 32, 0, stream>>>(
        aggF, cnt, vfW1p, vf_b1, vfW2p, vf_b2, outFeat, N);

    pos_kernel<<<(N * 3 * CCH + 255) / 256, 256, 0, stream>>>(
        node_pos, aggV, cnt, outPos, N);
}
